// PointNetSetAbstractionMsgWithOriginalGraph_76398878261705
// MI455X (gfx1250) — matrix operand order repaired and checked by reading, not compiled, not run on hardware
//
#include <hip/hip_runtime.h>
#include <hip/hip_bf16.h>
#include <math.h>

// ---------------------------------------------------------------------------
// Problem constants (from reference)
// ---------------------------------------------------------------------------
constexpr int BATCH   = 4;
constexpr int NPTS    = 4096;   // N
constexpr int NSAMPLE = 1024;   // S (fps points)
constexpr int INCH    = 64;
constexpr int KORIG   = 24;
constexpr int GRAPHC  = 128;    // GRAPH_OUT
constexpr int MAPC    = 320;    // MAP_OUT
constexpr int OUTCH   = 960;    // 640 + 320
constexpr float EPSBN = 1e-5f;

typedef __attribute__((ext_vector_type(16))) __bf16        v16bf;
typedef __attribute__((ext_vector_type(8)))  float         v8f;
typedef __attribute__((ext_vector_type(4)))  unsigned int  u32x4;

union FragBF { unsigned int u[8]; v16bf v; };

__device__ inline unsigned short f2bf(float f) {
    unsigned int u = __float_as_uint(f);
    u += 0x7FFFu + ((u >> 16) & 1u);          // round-to-nearest-even
    return (unsigned short)(u >> 16);
}

// ---------------------------------------------------------------------------
// Farthest point sampling: one workgroup per batch; min-dist kept in regs,
// argmax via LDS tree (tie -> smaller index to match jnp.argmax).
// ---------------------------------------------------------------------------
__global__ void fps_kernel(const float* __restrict__ xyz, int* __restrict__ fpsIdx)
{
    __shared__ float sd[256];
    __shared__ int   si[256];
    __shared__ int   sfar;
    const int b = blockIdx.x;
    const int t = threadIdx.x;
    const float* xb = xyz + (size_t)b * 3 * NPTS;
    float dmin[NPTS / 256];
    #pragma unroll
    for (int i = 0; i < NPTS / 256; ++i) dmin[i] = 1e10f;
    if (t == 0) sfar = 0;
    __syncthreads();
    for (int s = 0; s < NSAMPLE; ++s) {
        int far = sfar;
        if (t == 0) fpsIdx[b * NSAMPLE + s] = far;
        float cx = xb[far], cy = xb[NPTS + far], cz = xb[2 * NPTS + far];
        float bmax = -1.0f; int bidx = 0;
        #pragma unroll
        for (int i = 0; i < NPTS / 256; ++i) {
            int n = t + i * 256;
            float dx = xb[n] - cx, dy = xb[NPTS + n] - cy, dz = xb[2 * NPTS + n] - cz;
            float d = dx * dx + dy * dy + dz * dz;
            dmin[i] = fminf(dmin[i], d);
            if (dmin[i] > bmax) { bmax = dmin[i]; bidx = n; }
        }
        sd[t] = bmax; si[t] = bidx;
        __syncthreads();
        for (int off = 128; off > 0; off >>= 1) {
            if (t < off) {
                if (sd[t + off] > sd[t] ||
                    (sd[t + off] == sd[t] && si[t + off] < si[t])) {
                    sd[t] = sd[t + off]; si[t] = si[t + off];
                }
            }
            __syncthreads();
        }
        if (t == 0) sfar = si[0];
        __syncthreads();
    }
}

// ---------------------------------------------------------------------------
// Ball query: first `ns` indices (ascending) with sqdist <= r^2; pad w/ first.
// ---------------------------------------------------------------------------
__global__ void ball_query(const float* __restrict__ xyz, const int* __restrict__ fpsIdx,
                           int* __restrict__ gidx, float r2, int ns)
{
    int q = blockIdx.x * blockDim.x + threadIdx.x;
    if (q >= BATCH * NSAMPLE) return;
    int b = q / NSAMPLE;
    const float* xb = xyz + (size_t)b * 3 * NPTS;
    int ci = fpsIdx[q];
    float cx = xb[ci], cy = xb[NPTS + ci], cz = xb[2 * NPTS + ci];
    int* g = gidx + (size_t)q * ns;
    int cnt = 0;
    for (int n = 0; n < NPTS && cnt < ns; ++n) {
        float dx = xb[n] - cx, dy = xb[NPTS + n] - cy, dz = xb[2 * NPTS + n] - cz;
        if (dx * dx + dy * dy + dz * dz <= r2) g[cnt++] = n;
    }
    int first = (cnt > 0) ? g[0] : 0;
    for (; cnt < ns; ++cnt) g[cnt] = first;
}

// ---------------------------------------------------------------------------
// Build grouped features (M x 96 bf16, feature-last): [pts(64), dxyz(3), 0-pad]
// ---------------------------------------------------------------------------
__global__ void group_gather(const float* __restrict__ xyz, const float* __restrict__ points,
                             const int* __restrict__ fpsIdx, const int* __restrict__ gidx,
                             unsigned short* __restrict__ Xg, int ns)
{
    int m = blockIdx.x * blockDim.x + threadIdx.x;
    int M = BATCH * NSAMPLE * ns;
    if (m >= M) return;
    int q = m / ns;
    int b = q / NSAMPLE;
    int n  = gidx[(size_t)q * ns + (m % ns)];
    int ci = fpsIdx[q];
    const float* pb = points + (size_t)b * INCH * NPTS;
    const float* xb = xyz + (size_t)b * 3 * NPTS;
    unsigned short* row = Xg + (size_t)m * 96;
    for (int c = 0; c < INCH; ++c) row[c] = f2bf(pb[c * NPTS + n]);
    row[64] = f2bf(xb[n] - xb[ci]);
    row[65] = f2bf(xb[NPTS + n] - xb[NPTS + ci]);
    row[66] = f2bf(xb[2 * NPTS + n] - xb[2 * NPTS + ci]);
    for (int c = 67; c < 96; ++c) row[c] = 0;
}

// ---------------------------------------------------------------------------
// Convert & zero-pad weights fp32 (oc x cin) -> bf16 (oc x kpad)
// ---------------------------------------------------------------------------
__global__ void convert_pad_w(const float* __restrict__ W, unsigned short* __restrict__ Wp,
                              int oc, int cin, int kpad)
{
    int i = blockIdx.x * blockDim.x + threadIdx.x;
    if (i >= oc * kpad) return;
    int r = i / kpad, c = i % kpad;
    Wp[i] = (c < cin) ? f2bf(W[r * cin + c]) : (unsigned short)0;
}

__global__ void zero_f32(float* p, int n)
{
    int i = blockIdx.x * blockDim.x + threadIdx.x;
    if (i < n) p[i] = 0.0f;
}

// ---------------------------------------------------------------------------
// bf16 WMMA GEMM, register-blocked over OC, K fully unrolled (templated):
//   Y[m,oc] = sum_k W[oc,k]*X[m,k] (+bias), fused BN statistics.
// One wave computes a 16(M) x 64(OC) strip: 4 accumulators sharing one
// B-fragment per K-step. Compile-time K lets the scheduler hoist all
// fragment loads (constant offsets) and issue WMMAs against in-flight loads
// instead of draining loadcnt before each WMMA.
// Grid: (M/128, OC/64). M %128==0, OC %64==0.
// A-frag (16-bit A 16x32): lanes 0-15 K{0-7,16-23}, lanes 16-31 K{8-15,24-31}
// B-frag (16-bit B 32x16): col = lane&15, lanes 0-15 K{0-7,16-23}, lanes 16-31 K{8-15,24-31}
// ---------------------------------------------------------------------------
constexpr int OCT = 4;   // OC tiles per wave

template <int KT>        // K = KT*32; network uses KT in {2, 3, 4}
__global__ void gemm_bf16_wmma(const unsigned short* __restrict__ W,
                               const unsigned short* __restrict__ X,
                               const float* __restrict__ bias,
                               float* __restrict__ Y,
                               float* __restrict__ sums,
                               float* __restrict__ sumsq,
                               int M, int OC)
{
    constexpr int K = KT * 32;
    __shared__ float bsum[16 * OCT];
    __shared__ float bsum2[16 * OCT];
    const int lane = threadIdx.x & 31;
    const int wave = threadIdx.x >> 5;
    const int hi = lane >> 4;
    const int lo = lane & 15;
    const int mBase  = (blockIdx.x * 8 + wave) * 16;
    const int ocBase = blockIdx.y * (16 * OCT);

    if (threadIdx.x < 16 * OCT) { bsum[threadIdx.x] = 0.0f; bsum2[threadIdx.x] = 0.0f; }
    __syncthreads();

    const unsigned short* wrow = W + (size_t)(ocBase + lo) * K;
    const unsigned short* xrow = X + (size_t)(mBase + lo) * K;
    constexpr size_t wstride = (size_t)16 * K;   // one OC tile of A rows

    v8f acc[OCT];
    #pragma unroll
    for (int t = 0; t < OCT; ++t)
        #pragma unroll
        for (int i = 0; i < 8; ++i) acc[t][i] = 0.0f;

    #pragma unroll
    for (int kt = 0; kt < KT; ++kt) {
        const int k = kt * 32;
        FragBF b;
        const int kb = k + hi * 8;
        *(u32x4*)&b.u[0] = *(const u32x4*)(xrow + kb);
        *(u32x4*)&b.u[4] = *(const u32x4*)(xrow + kb + 16);
        const int ka = k + hi * 8;
        #pragma unroll
        for (int t = 0; t < OCT; ++t) {
            FragBF a;
            *(u32x4*)&a.u[0] = *(const u32x4*)(wrow + t * wstride + ka);
            *(u32x4*)&a.u[4] = *(const u32x4*)(wrow + t * wstride + ka + 16);
            acc[t] = __builtin_amdgcn_wmma_f32_16x16x32_bf16(false, a.v, false, b.v,
                                                             (short)0, acc[t], false, false);
        }
    }

    const int m = mBase + lo;
    #pragma unroll
    for (int t = 0; t < OCT; ++t) {
        #pragma unroll
        for (int v = 0; v < 8; ++v) {
            const int cLoc = 16 * t + v + 8 * hi;      // channel within block strip
            const int oc = ocBase + cLoc;
            float val = acc[t][v] + (bias ? bias[oc] : 0.0f);
            Y[(size_t)m * OC + oc] = val;
            // reduce over the 16 columns of this tile (xor stays inside hi-group)
            float s1 = val, s2 = val * val;
            #pragma unroll
            for (int off = 1; off < 16; off <<= 1) {
                s1 += __shfl_xor(s1, off, 32);
                s2 += __shfl_xor(s2, off, 32);
            }
            if (lo == 0) {
                atomicAdd(&bsum[cLoc], s1);    // LDS ds_add_f32
                atomicAdd(&bsum2[cLoc], s2);
            }
        }
    }
    __syncthreads();
    if (threadIdx.x < 16 * OCT) {
        atomicAdd(&sums[ocBase + threadIdx.x],  bsum[threadIdx.x]);
        atomicAdd(&sumsq[ocBase + threadIdx.x], bsum2[threadIdx.x]);
    }
}

// ---------------------------------------------------------------------------
// BN: scale = gamma*rsqrt(var+eps), shift = beta - mean*scale
// ---------------------------------------------------------------------------
__global__ void bn_finalize(const float* __restrict__ sums, const float* __restrict__ sumsq,
                            const float* __restrict__ gamma, const float* __restrict__ beta,
                            float* __restrict__ scl, float* __restrict__ shf,
                            int C, float invM)
{
    int c = blockIdx.x * blockDim.x + threadIdx.x;
    if (c >= C) return;
    float mean = sums[c] * invM;
    float var  = sumsq[c] * invM - mean * mean;
    float s = gamma[c] * rsqrtf(var + EPSBN);
    scl[c] = s;
    shf[c] = beta[c] - mean * s;
}

// BN + activation (slope=0 -> relu), fp32 -> bf16 for next GEMM
__global__ void bn_act_to_bf16(const float* __restrict__ Y, const float* __restrict__ scl,
                               const float* __restrict__ shf, unsigned short* __restrict__ X,
                               size_t total, int C, float slope)
{
    size_t i = (size_t)blockIdx.x * blockDim.x + threadIdx.x;
    if (i >= total) return;
    int c = (int)(i % C);
    float v = scl[c] * Y[i] + shf[c];
    v = v > 0.0f ? v : v * slope;
    X[i] = f2bf(v);
}

// BN + relu + max over nsample, write output slice (B, OUTCH, S)
__global__ void bn_relu_max_out(const float* __restrict__ Y, const float* __restrict__ scl,
                                const float* __restrict__ shf, float* __restrict__ out,
                                int OC, int ns, int chOff)
{
    int t = blockIdx.x * blockDim.x + threadIdx.x;
    if (t >= BATCH * NSAMPLE * OC) return;
    int oc = t % OC;
    int s  = (t / OC) % NSAMPLE;
    int b  = t / (OC * NSAMPLE);
    size_t base = ((size_t)(b * NSAMPLE + s)) * ns;
    float sc = scl[oc], sh = shf[oc];
    float mx = -3.4e38f;
    for (int j = 0; j < ns; ++j) {
        float v = sc * Y[(base + j) * OC + oc] + sh;
        v = v > 0.0f ? v : 0.0f;
        mx = fmaxf(mx, v);
    }
    out[(size_t)b * OUTCH * NSAMPLE + (size_t)(chOff + oc) * NSAMPLE + s] = mx;
}

// ---------------------------------------------------------------------------
// kNN (k=24, excluding self): per-point insertion into 25-slot sorted list
// ---------------------------------------------------------------------------
__global__ void knn_kernel(const float* __restrict__ xyz, int* __restrict__ knn)
{
    int t = blockIdx.x * blockDim.x + threadIdx.x;
    if (t >= BATCH * NPTS) return;
    int b = t / NPTS, n = t % NPTS;
    const float* xb = xyz + (size_t)b * 3 * NPTS;
    float px = xb[n], py = xb[NPTS + n], pz = xb[2 * NPTS + n];
    float bd[KORIG + 1];
    int   bi[KORIG + 1];
    for (int i = 0; i <= KORIG; ++i) { bd[i] = 3.4e38f; bi[i] = 0; }
    for (int m = 0; m < NPTS; ++m) {
        float dx = xb[m] - px, dy = xb[NPTS + m] - py, dz = xb[2 * NPTS + m] - pz;
        float d = dx * dx + dy * dy + dz * dz;
        if (d < bd[KORIG]) {
            int p = KORIG;
            while (p > 0 && bd[p - 1] > d) { bd[p] = bd[p - 1]; bi[p] = bi[p - 1]; --p; }
            bd[p] = d; bi[p] = m;
        }
    }
    for (int k = 0; k < KORIG; ++k) knn[(size_t)t * KORIG + k] = bi[k + 1];
}

// Edge features (Me x 128 bf16): [x_c(64), x_nb - x_c(64)]
__global__ void edge_gather(const float* __restrict__ points, const int* __restrict__ knn,
                            unsigned short* __restrict__ Xe)
{
    int t = blockIdx.x * blockDim.x + threadIdx.x;   // (b*N+n)*K + k
    if (t >= BATCH * NPTS * KORIG) return;
    int bn = t / KORIG;
    int b = bn / NPTS, n = bn % NPTS;
    int nb = knn[t];
    const float* pb = points + (size_t)b * INCH * NPTS;
    unsigned short* row = Xe + (size_t)t * (2 * INCH);
    for (int c = 0; c < INCH; ++c) {
        float xc = pb[c * NPTS + n];
        float xn = pb[c * NPTS + nb];
        row[c] = f2bf(xc);
        row[INCH + c] = f2bf(xn - xc);
    }
}

// BN + leaky(0.2) + max over k -> graph_feat (B*N x 128 bf16, feature-last)
__global__ void bn_lrelu_max_graph(const float* __restrict__ Ye, const float* __restrict__ scl,
                                   const float* __restrict__ shf, unsigned short* __restrict__ GF)
{
    int t = blockIdx.x * blockDim.x + threadIdx.x;   // (b*N+n)*128 + oc
    if (t >= BATCH * NPTS * GRAPHC) return;
    int oc = t % GRAPHC;
    int bn = t / GRAPHC;
    float sc = scl[oc], sh = shf[oc];
    float mx = -3.4e38f;
    size_t base = (size_t)bn * KORIG;
    for (int k = 0; k < KORIG; ++k) {
        float v = sc * Ye[(base + k) * GRAPHC + oc] + sh;
        v = v > 0.0f ? v : 0.2f * v;
        mx = fmaxf(mx, v);
    }
    GF[t] = f2bf(mx);
}

// map BN + leaky + gather at fps indices -> output channels 640..959
__global__ void map_apply_gather(const float* __restrict__ Ym, const float* __restrict__ scl,
                                 const float* __restrict__ shf, const int* __restrict__ fpsIdx,
                                 float* __restrict__ out)
{
    int t = blockIdx.x * blockDim.x + threadIdx.x;   // (b*MAPC + c)*S + s
    if (t >= BATCH * MAPC * NSAMPLE) return;
    int s = t % NSAMPLE;
    int c = (t / NSAMPLE) % MAPC;
    int b = t / (NSAMPLE * MAPC);
    int n = fpsIdx[b * NSAMPLE + s];
    float v = scl[c] * Ym[((size_t)(b * NPTS + n)) * MAPC + c] + shf[c];
    v = v > 0.0f ? v : 0.2f * v;
    out[(size_t)b * OUTCH * NSAMPLE + (size_t)(640 + c) * NSAMPLE + s] = v;
}

// ---------------------------------------------------------------------------
// GEMM dispatch over compile-time K
// ---------------------------------------------------------------------------
static void launch_gemm(int K, const unsigned short* W, const unsigned short* X,
                        const float* bias, float* Y, float* sums, float* sumsq,
                        int M, int OC, hipStream_t stream)
{
    dim3 grid(M / 128, OC / 64);
    switch (K / 32) {
    case 2: gemm_bf16_wmma<2><<<grid, 256, 0, stream>>>(W, X, bias, Y, sums, sumsq, M, OC); break;
    case 3: gemm_bf16_wmma<3><<<grid, 256, 0, stream>>>(W, X, bias, Y, sums, sumsq, M, OC); break;
    case 4: gemm_bf16_wmma<4><<<grid, 256, 0, stream>>>(W, X, bias, Y, sums, sumsq, M, OC); break;
    default: break; // network only uses K in {64, 96, 128}
    }
}

// ---------------------------------------------------------------------------
// Host orchestration
// ---------------------------------------------------------------------------
extern "C" void kernel_launch(void* const* d_in, const int* in_sizes, int n_in,
                              void* d_out, int out_size, void* d_ws, size_t ws_size,
                              hipStream_t stream)
{
    (void)in_sizes; (void)n_in; (void)out_size; (void)ws_size;
    const float* xyz    = (const float*)d_in[0];
    const float* points = (const float*)d_in[1];

    struct LayerP { const float *W, *b, *g, *bt; int oc, cin, kpad; };
    static const int mlp_oc[3][3] = {{64, 64, 128}, {128, 128, 256}, {128, 128, 256}};
    LayerP L[3][3];
    int pi = 2;
    for (int i = 0; i < 3; ++i) {
        int last = 67;
        for (int l = 0; l < 3; ++l) {
            L[i][l].W  = (const float*)d_in[pi++];
            L[i][l].b  = (const float*)d_in[pi++];
            L[i][l].g  = (const float*)d_in[pi++];
            L[i][l].bt = (const float*)d_in[pi++];
            L[i][l].oc = mlp_oc[i][l];
            L[i][l].cin = last;
            L[i][l].kpad = (last == 67) ? 96 : last;   // pad 67 -> 96 (K%32==0)
            last = mlp_oc[i][l];
        }
    }
    const float* We = (const float*)d_in[pi++];
    const float* ge = (const float*)d_in[pi++];
    const float* be = (const float*)d_in[pi++];
    const float* Wm = (const float*)d_in[pi++];
    const float* gm = (const float*)d_in[pi++];
    const float* bm = (const float*)d_in[pi++];

    // ---- workspace carve-up (256B aligned regions) ----
    char* ws = (char*)d_ws;
    size_t off = 0;
    auto carve = [&](size_t bytes) -> char* {
        char* p = ws + off;
        off += (bytes + 255) & ~(size_t)255;
        return p;
    };
    int*            fpsIdx = (int*)carve((size_t)BATCH * NSAMPLE * 4);
    int*            gidx   = (int*)carve((size_t)BATCH * NSAMPLE * 64 * 4);
    int*            knnIdx = (int*)carve((size_t)BATCH * NPTS * KORIG * 4);
    unsigned short* wpad   = (unsigned short*)carve((size_t)MAPC * 128 * 2);
    float*          sums   = (float*)carve(512 * 4);
    float*          sumsq  = (float*)carve(512 * 4);
    float*          sclA   = (float*)carve(512 * 4);
    float*          shfA   = (float*)carve(512 * 4);
    unsigned short* GF     = (unsigned short*)carve((size_t)BATCH * NPTS * GRAPHC * 2);
    float*          Ym     = (float*)carve((size_t)BATCH * NPTS * MAPC * 4);
    unsigned short* bufX   = (unsigned short*)carve((size_t)50331648 * 2); // max M*K bf16
    float*          bufY   = (float*)carve((size_t)67108864 * 4);          // max M*OC f32

    static const int   nsArr[3]    = {16, 32, 64};
    static const float r2Arr[3]    = {0.01f, 0.04f, 0.16f};
    static const int   chOffArr[3] = {0, 128, 384};

    fps_kernel<<<BATCH, 256, 0, stream>>>(xyz, fpsIdx);

    // ---- three MSG scales ----
    for (int i = 0; i < 3; ++i) {
        int ns = nsArr[i];
        int M  = BATCH * NSAMPLE * ns;
        ball_query<<<(BATCH * NSAMPLE + 255) / 256, 256, 0, stream>>>(xyz, fpsIdx, gidx, r2Arr[i], ns);
        group_gather<<<(M + 255) / 256, 256, 0, stream>>>(xyz, points, fpsIdx, gidx, bufX, ns);
        for (int l = 0; l < 3; ++l) {
            const LayerP& P = L[i][l];
            int K = P.kpad, OC = P.oc;
            convert_pad_w<<<(OC * K + 255) / 256, 256, 0, stream>>>(P.W, wpad, OC, P.cin, K);
            zero_f32<<<2, 256, 0, stream>>>(sums, 512);
            zero_f32<<<2, 256, 0, stream>>>(sumsq, 512);
            launch_gemm(K, wpad, bufX, P.b, bufY, sums, sumsq, M, OC, stream);
            bn_finalize<<<(OC + 63) / 64, 64, 0, stream>>>(sums, sumsq, P.g, P.bt, sclA, shfA, OC,
                                                           1.0f / (float)M);
            if (l < 2) {
                size_t total = (size_t)M * OC;
                bn_act_to_bf16<<<(unsigned)((total + 255) / 256), 256, 0, stream>>>(
                    bufY, sclA, shfA, bufX, total, OC, 0.0f);
            } else {
                int tot = BATCH * NSAMPLE * OC;
                bn_relu_max_out<<<(tot + 255) / 256, 256, 0, stream>>>(
                    bufY, sclA, shfA, (float*)d_out, OC, ns, chOffArr[i]);
            }
        }
    }

    // ---- edge-conv branch ----
    knn_kernel<<<(BATCH * NPTS + 255) / 256, 256, 0, stream>>>(xyz, knnIdx);
    {
        int Me = BATCH * NPTS * KORIG;
        edge_gather<<<(Me + 255) / 256, 256, 0, stream>>>(points, knnIdx, bufX);
        convert_pad_w<<<(GRAPHC * 128 + 255) / 256, 256, 0, stream>>>(We, wpad, GRAPHC, 2 * INCH, 128);
        zero_f32<<<2, 256, 0, stream>>>(sums, 512);
        zero_f32<<<2, 256, 0, stream>>>(sumsq, 512);
        launch_gemm(128, wpad, bufX, nullptr, bufY, sums, sumsq, Me, GRAPHC, stream);
        bn_finalize<<<2, 64, 0, stream>>>(sums, sumsq, ge, be, sclA, shfA, GRAPHC, 1.0f / (float)Me);
        int tg = BATCH * NPTS * GRAPHC;
        bn_lrelu_max_graph<<<(tg + 255) / 256, 256, 0, stream>>>(bufY, sclA, shfA, GF);
    }
    // ---- map layer + gather ----
    {
        int Mm = BATCH * NPTS;
        convert_pad_w<<<(MAPC * 128 + 255) / 256, 256, 0, stream>>>(Wm, wpad, MAPC, GRAPHC, 128);
        zero_f32<<<2, 256, 0, stream>>>(sums, 512);
        zero_f32<<<2, 256, 0, stream>>>(sumsq, 512);
        launch_gemm(128, wpad, GF, nullptr, Ym, sums, sumsq, Mm, MAPC, stream);
        bn_finalize<<<(MAPC + 63) / 64, 64, 0, stream>>>(sums, sumsq, gm, bm, sclA, shfA, MAPC,
                                                         1.0f / (float)Mm);
        int tm = BATCH * MAPC * NSAMPLE;
        map_apply_gather<<<(tm + 255) / 256, 256, 0, stream>>>(Ym, sclA, shfA, fpsIdx,
                                                               (float*)d_out);
    }
}
